// LightPointCloudNet_44641890075340
// MI455X (gfx1250) — compile-verified, matrix operand order checked by reading, not run on hardware
//
#include <hip/hip_runtime.h>
#include <hip/hip_bf16.h>
#include <cstdint>

// ---------------------------------------------------------------------------
// Problem constants (reference shapes are fixed)
// ---------------------------------------------------------------------------
constexpr int BATCH = 4;
constexpr int LPTS  = 8192;   // points per cloud
constexpr int DIN   = 6;

typedef __attribute__((ext_vector_type(16))) _Float16 v16h;
typedef __attribute__((ext_vector_type(8)))  float    v8f;

union U16H { v16h v; uint32_t u[8]; };

// f16 A/B operand pair layout for V_WMMA_*_16x16x32_f16 (per ISA 7.12.2):
// lanes 0-15 hold K=0..7 (VGPR0..3) and K=16..23 (VGPR4..7);
// lanes 16-31 hold K=8..15 and K=24..31. Each VGPR packs two adjacent K.
__device__ __forceinline__ int pair_base(int half16, int j) {
  return ((j < 4) ? 0 : 16) + half16 * 8 + (j & 3) * 2;
}

// ---------------------------------------------------------------------------
// 1. Input normalization: center, scale by max radius, transpose -> (N,6,L)
// ---------------------------------------------------------------------------
__global__ void normalize_kernel(const float* __restrict__ x, float* __restrict__ xT) {
  const int n = blockIdx.x;
  const int tid = threadIdx.x;
  __shared__ float red[256];
  __shared__ float meanv[DIN];
  __shared__ float invd;
  const float* xb = x + (size_t)n * LPTS * DIN;

  for (int d = 0; d < DIN; ++d) {
    float s = 0.f;
    for (int l = tid; l < LPTS; l += 256) s += xb[(size_t)l * DIN + d];
    red[tid] = s; __syncthreads();
    for (int off = 128; off >= 1; off >>= 1) {
      if (tid < off) red[tid] += red[tid + off];
      __syncthreads();
    }
    if (tid == 0) meanv[d] = red[0] / (float)LPTS;
    __syncthreads();
  }
  float mx = 0.f;
  for (int l = tid; l < LPTS; l += 256) {
    float s2 = 0.f;
    for (int d = 0; d < DIN; ++d) {
      float v = xb[(size_t)l * DIN + d] - meanv[d];
      s2 += v * v;
    }
    mx = fmaxf(mx, sqrtf(s2));
  }
  red[tid] = mx; __syncthreads();
  for (int off = 128; off >= 1; off >>= 1) {
    if (tid < off) red[tid] = fmaxf(red[tid], red[tid + off]);
    __syncthreads();
  }
  if (tid == 0) invd = 1.f / (red[0] + 1e-6f);
  __syncthreads();
  const float inv = invd;
  for (int l = tid; l < LPTS; l += 256)
    for (int d = 0; d < DIN; ++d)
      xT[((size_t)n * DIN + d) * LPTS + l] = (xb[(size_t)l * DIN + d] - meanv[d]) * inv;
}

// ---------------------------------------------------------------------------
// 2. 1x1 conv: out(N,O,L) = W(O,C) * in(N,C,L) + b
// ---------------------------------------------------------------------------
__global__ void conv1x1_kernel(const float* __restrict__ in, const float* __restrict__ W,
                               const float* __restrict__ bias, float* __restrict__ out,
                               int C, int O) {
  const int n = blockIdx.y;
  const size_t i = (size_t)blockIdx.x * blockDim.x + threadIdx.x;
  const int o = (int)(i / LPTS);
  const int l = (int)(i % LPTS);
  if (o >= O) return;
  float acc = bias[o];
  const float* ip = in + (size_t)n * C * LPTS + l;
  const float* wp = W + (size_t)o * C;
  for (int c = 0; c < C; ++c) acc += wp[c] * ip[(size_t)c * LPTS];
  out[((size_t)n * O + o) * LPTS + l] = acc;
}

// ---------------------------------------------------------------------------
// 3. GroupNorm(num_groups=1): per-batch stats over C*L, then affine (+ReLU)
// ---------------------------------------------------------------------------
__global__ void gn_stats_kernel(const float* __restrict__ x, float* __restrict__ stats, int CL) {
  const int n = blockIdx.x;
  const int tid = threadIdx.x;
  __shared__ float s1[256], s2[256];
  const float* xb = x + (size_t)n * CL;
  float a = 0.f, b = 0.f;
  for (int i = tid; i < CL; i += 256) { float v = xb[i]; a += v; b += v * v; }
  s1[tid] = a; s2[tid] = b; __syncthreads();
  for (int off = 128; off >= 1; off >>= 1) {
    if (tid < off) { s1[tid] += s1[tid + off]; s2[tid] += s2[tid + off]; }
    __syncthreads();
  }
  if (tid == 0) {
    float m = s1[0] / (float)CL;
    float var = s2[0] / (float)CL - m * m;
    stats[n * 2]     = m;
    stats[n * 2 + 1] = rsqrtf(var + 1e-3f);   // EPS_GN
  }
}

__global__ void gn_apply_kernel(float* __restrict__ x, const float* __restrict__ stats,
                                const float* __restrict__ g, const float* __restrict__ b,
                                int C, int L, int relu) {
  const size_t i = (size_t)blockIdx.x * blockDim.x + threadIdx.x;
  const size_t total = (size_t)BATCH * C * L;
  if (i >= total) return;
  const int n = (int)(i / ((size_t)C * L));
  const int c = (int)((i / L) % C);
  const float m = stats[n * 2], r = stats[n * 2 + 1];
  float v = (x[i] - m) * r * g[c] + b[c];
  if (relu) v = fmaxf(v, 0.f);
  x[i] = v;
}

// ---------------------------------------------------------------------------
// 4. Max-pool over L: out(N,C) = max_l in(N,C,L). one block per (n,c)
// ---------------------------------------------------------------------------
__global__ void maxpool_kernel(const float* __restrict__ in, float* __restrict__ out) {
  const int nc = blockIdx.x;
  const int tid = threadIdx.x;
  __shared__ float red[256];
  const float* ip = in + (size_t)nc * LPTS;
  float m = -__builtin_inff();
  for (int l = tid; l < LPTS; l += 256) m = fmaxf(m, ip[l]);
  red[tid] = m; __syncthreads();
  for (int off = 128; off >= 1; off >>= 1) {
    if (tid < off) red[tid] = fmaxf(red[tid], red[tid + off]);
    __syncthreads();
  }
  if (tid == 0) out[nc] = red[0];
}

// ---------------------------------------------------------------------------
// 5. Fully connected: out(N,O) = in(N,F) @ W(O,F)^T + b   (N=4 -> tiny)
// ---------------------------------------------------------------------------
__global__ void fc_kernel(const float* __restrict__ in, const float* __restrict__ W,
                          const float* __restrict__ b, float* __restrict__ out,
                          int F, int O) {
  const int i = blockIdx.x * blockDim.x + threadIdx.x;
  if (i >= BATCH * O) return;
  const int n = i / O, o = i % O;
  float acc = b[o];
  const float* ip = in + (size_t)n * F;
  const float* wp = W + (size_t)o * F;
  for (int f = 0; f < F; ++f) acc += ip[f] * wp[f];
  out[i] = acc;
}

__global__ void fc_clip_kernel(const float* __restrict__ in, const float* __restrict__ W,
                               const float* __restrict__ b, float* __restrict__ out, int F) {
  const int n = blockIdx.x * blockDim.x + threadIdx.x;
  if (n >= BATCH) return;
  float acc = b[0];
  for (int f = 0; f < F; ++f) acc += in[(size_t)n * F + f] * W[f];
  out[n] = fminf(fmaxf(acc, -100.f), 100.f);
}

// ---------------------------------------------------------------------------
// 6. TNet tail: trans = 0.1*tanh(fc3) + I6  (N,6,6)
// ---------------------------------------------------------------------------
__global__ void tnet_final_kernel(const float* __restrict__ fc3, float* __restrict__ trans) {
  const int i = blockIdx.x * blockDim.x + threadIdx.x;
  if (i >= BATCH * 36) return;
  const int rc = i % 36;
  const int r = rc / 6, c = rc % 6;
  trans[i] = 0.1f * tanhf(fc3[i]) + (r == c ? 1.f : 0.f);
}

// x2[n,i,l] = sum_j trans[n,i,j] * xT[n,j,l]
__global__ void apply_trans_kernel(const float* __restrict__ trans, const float* __restrict__ xT,
                                   float* __restrict__ out) {
  const size_t idx = (size_t)blockIdx.x * blockDim.x + threadIdx.x;
  if (idx >= (size_t)BATCH * DIN * LPTS) return;
  const int n = (int)(idx / ((size_t)DIN * LPTS));
  const int i = (int)((idx / LPTS) % DIN);
  const int l = (int)(idx % LPTS);
  float acc = 0.f;
  const float* tp = trans + (size_t)n * 36 + (size_t)i * 6;
  const float* xp = xT + (size_t)n * DIN * LPTS + l;
  #pragma unroll
  for (int j = 0; j < DIN; ++j) acc += tp[j] * xp[(size_t)j * LPTS];
  out[idx] = acc;
}

// ---------------------------------------------------------------------------
// 7. Attention operand packing (f32 -> f16)
//    q,k: (N,16,L) -> (N,L,32) padded with zeros (K dim padded to 32)
//    v  : (N,128,L) -> f16 same layout
// ---------------------------------------------------------------------------
__global__ void pack_qk_kernel(const float* __restrict__ q, _Float16* __restrict__ qh) {
  const size_t i = (size_t)blockIdx.x * blockDim.x + threadIdx.x;   // over N*L*32
  if (i >= (size_t)BATCH * LPTS * 32) return;
  const int c = (int)(i % 32);
  const size_t nl = i / 32;
  const int n = (int)(nl / LPTS);
  const int l = (int)(nl % LPTS);
  qh[i] = (c < 16) ? (_Float16)q[((size_t)n * 16 + c) * LPTS + l] : (_Float16)0.f;
}

__global__ void pack_v_kernel(const float* __restrict__ v, _Float16* __restrict__ vh) {
  const size_t i = (size_t)blockIdx.x * blockDim.x + threadIdx.x;
  if (i >= (size_t)BATCH * 128 * LPTS) return;
  vh[i] = (_Float16)v[i];
}

// ---------------------------------------------------------------------------
// 8. Flash attention (the WMMA kernel).
//    One wave32 owns a 16-query tile; loops keys in chunks of 64:
//      S    = Q(16x32) x K^T          -> 4x v_wmma_f32_16x16x32_f16
//      online softmax: row-max via shfl_xor (16-lane halves),
//                      row-sum via P x ones-matrix WMMA (2x v_wmma)
//      O   += P(16x64) x V^T          -> 16x v_wmma (128 channels)
//    out = gamma * (O / rowsum) + x   (residual), written to (N,128,L)
// ---------------------------------------------------------------------------
__global__ void __launch_bounds__(128)
attn_flash_kernel(const _Float16* __restrict__ qh,   // (N, L, 32)
                  const _Float16* __restrict__ kh,   // (N, L, 32)
                  const _Float16* __restrict__ vh,   // (N, 128, L)
                  const float*    __restrict__ xin,  // (N, 128, L)
                  const float*    __restrict__ gammap,
                  float*          __restrict__ out)  // (N, 128, L)
{
  const int lane  = threadIdx.x & 31;
  const int wave  = threadIdx.x >> 5;
  const int n     = blockIdx.y;
  const int qtile = blockIdx.x * 4 + wave;
  const int qbase = qtile * 16;
  const int h16   = lane >> 4;   // which 16-lane half
  const int ncol  = lane & 15;
  const float gamma = *gammap;

  __shared__ _Float16 Plds[4][16][64];   // per-wave P staging (D->A transpose)

  // Q operand: A matrix 16x32, row = query, K = channel (padded)
  U16H Qa;
  {
    const uint32_t* qrow =
        (const uint32_t*)(qh + ((size_t)n * LPTS + (qbase + ncol)) * 32);
    #pragma unroll
    for (int j = 0; j < 8; ++j) Qa.u[j] = qrow[pair_base(h16, j) >> 1];
  }

  // all-ones B operand: rowsum(P) = P x Ones via the matrix pipe
  U16H OnesB;
  #pragma unroll
  for (int j = 0; j < 8; ++j) OnesB.u[j] = 0x3C003C00u;   // two f16 1.0

  float rmax[8], rsum[8];
  v8f acc[8];
  #pragma unroll
  for (int r = 0; r < 8; ++r) { rmax[r] = -__builtin_inff(); rsum[r] = 0.f; }
  #pragma unroll
  for (int t = 0; t < 8; ++t) acc[t] = (v8f){0.f, 0.f, 0.f, 0.f, 0.f, 0.f, 0.f, 0.f};
  const v8f zero8 = (v8f){0.f, 0.f, 0.f, 0.f, 0.f, 0.f, 0.f, 0.f};

  for (int mbase = 0; mbase < LPTS; mbase += 64) {
    // K operands: B matrix 32x16, K = channel, N = key; 4 groups of 16 keys
    U16H Bk[4];
    #pragma unroll
    for (int g = 0; g < 4; ++g) {
      const uint32_t* krow =
          (const uint32_t*)(kh + ((size_t)n * LPTS + (mbase + g * 16 + ncol)) * 32);
      #pragma unroll
      for (int j = 0; j < 8; ++j) Bk[g].u[j] = krow[pair_base(h16, j) >> 1];
    }
    // keep the next K chunk ahead of the matrix pipe (global_prefetch_b8)
    if (mbase + 64 < LPTS)
      __builtin_prefetch(kh + ((size_t)n * LPTS + (mbase + 64 + lane * 2)) * 32, 0, 1);

    v8f S[4];
    #pragma unroll
    for (int g = 0; g < 4; ++g)
      S[g] = __builtin_amdgcn_wmma_f32_16x16x32_f16(false, Qa.v, false, Bk[g].v,
                                                    (short)0, zero8, false, false);

    // Online softmax: row max over 64 keys (4 regs + 4 shfl stages), rescale
    float p[4][8];
    float carr[8];
    #pragma unroll
    for (int r = 0; r < 8; ++r) {
      float mx = fmaxf(fmaxf(S[0][r], S[1][r]), fmaxf(S[2][r], S[3][r]));
      #pragma unroll
      for (int m = 8; m >= 1; m >>= 1) mx = fmaxf(mx, __shfl_xor(mx, m, 32));
      const float mnew = fmaxf(rmax[r], mx);
      const float corr = __expf(rmax[r] - mnew);
      rmax[r] = mnew;
      carr[r] = corr;
      #pragma unroll
      for (int g = 0; g < 4; ++g) p[g][r] = __expf(S[g][r] - mnew);
      #pragma unroll
      for (int t = 0; t < 8; ++t) acc[t][r] *= corr;
    }

    // D-layout -> A-layout transpose of P through LDS (intra-wave only)
    #pragma unroll
    for (int r = 0; r < 8; ++r) {
      #pragma unroll
      for (int g = 0; g < 4; ++g)
        Plds[wave][r + 8 * h16][g * 16 + ncol] = (_Float16)p[g][r];
    }
    U16H Pa0, Pa1;
    #pragma unroll
    for (int j = 0; j < 8; ++j) {
      const int pb = pair_base(h16, j);
      Pa0.u[j] = *(const uint32_t*)&Plds[wave][ncol][pb];
      Pa1.u[j] = *(const uint32_t*)&Plds[wave][ncol][32 + pb];
    }

    // row sums on the matrix pipe: st[r] = sum_k P[row,k] (replicated per lane)
    v8f st = __builtin_amdgcn_wmma_f32_16x16x32_f16(false, Pa0.v, false, OnesB.v,
                                                    (short)0, zero8, false, false);
    st = __builtin_amdgcn_wmma_f32_16x16x32_f16(false, Pa1.v, false, OnesB.v,
                                                (short)0, st, false, false);
    #pragma unroll
    for (int r = 0; r < 8; ++r) rsum[r] = rsum[r] * carr[r] + st[r];

    // O += P x V^T across 8 channel tiles; V rows stream straight from L2
    #pragma unroll
    for (int t = 0; t < 8; ++t) {
      U16H Vb0, Vb1;
      const _Float16* vrow =
          vh + ((size_t)n * 128 + (t * 16 + ncol)) * LPTS + mbase;
      #pragma unroll
      for (int j = 0; j < 8; ++j) {
        const int pb = pair_base(h16, j);
        Vb0.u[j] = *(const uint32_t*)(vrow + pb);
        Vb1.u[j] = *(const uint32_t*)(vrow + 32 + pb);
      }
      acc[t] = __builtin_amdgcn_wmma_f32_16x16x32_f16(false, Pa0.v, false, Vb0.v,
                                                      (short)0, acc[t], false, false);
      acc[t] = __builtin_amdgcn_wmma_f32_16x16x32_f16(false, Pa1.v, false, Vb1.v,
                                                      (short)0, acc[t], false, false);
    }
  }

  // Epilogue: normalize, residual, store
  #pragma unroll
  for (int r = 0; r < 8; ++r) {
    const float inv = 1.f / rsum[r];
    const int l = qbase + r + 8 * h16;
    #pragma unroll
    for (int t = 0; t < 8; ++t) {
      const int c = t * 16 + ncol;
      const size_t idx = ((size_t)n * 128 + c) * LPTS + l;
      out[idx] = gamma * acc[t][r] * inv + xin[idx];
    }
  }
}

// ---------------------------------------------------------------------------
// Host-side orchestration
// ---------------------------------------------------------------------------
static inline unsigned cdiv(size_t a, size_t b) { return (unsigned)((a + b - 1) / b); }

extern "C" void kernel_launch(void* const* d_in, const int* in_sizes, int n_in,
                              void* d_out, int out_size, void* d_ws, size_t ws_size,
                              hipStream_t stream) {
  (void)in_sizes; (void)n_in; (void)out_size; (void)ws_size;
  const float* X = (const float*)d_in[0];
  #define PARAM(i) ((const float*)d_in[i])
  // param index map (setup_inputs dict insertion order)
  enum {
    T_C1W = 1, T_C1B, T_N1G, T_N1B,
    T_C2W, T_C2B, T_N2G, T_N2B,
    T_C3W, T_C3B, T_N3G, T_N3B,
    T_F1W, T_F1B, T_N4G, T_N4B,
    T_F2W, T_F2B, T_N5G, T_N5B,
    T_F3W, T_F3B,
    C1W, C1B, N1G, N1B,
    C2W, C2B, N2G, N2B,
    C3W, C3B, N3G, N3B,
    A_QW, A_QB, A_KW, A_KB, A_VW, A_VB, A_GAMMA,
    F1W, F1B, N4G, N4B,
    F2W, F2B, N5G, N5B,
    F3W, F3B
  };

  // workspace carve-up
  char* wsp = (char*)d_ws;
  auto alloc = [&](size_t bytes) -> char* {
    char* p = wsp;
    wsp += (bytes + 255) & ~(size_t)255;
    return p;
  };
  float* xT     = (float*)alloc(sizeof(float) * BATCH * DIN * LPTS);
  float* x2     = (float*)alloc(sizeof(float) * BATCH * DIN * LPTS);
  float* h1     = (float*)alloc(sizeof(float) * BATCH * 32  * LPTS);
  float* h2     = (float*)alloc(sizeof(float) * BATCH * 64  * LPTS);
  float* h3     = (float*)alloc(sizeof(float) * BATCH * 256 * LPTS);  // reused (tnet 256, trunk 128)
  float* qbuf   = (float*)alloc(sizeof(float) * BATCH * 16  * LPTS);
  float* kbuf   = (float*)alloc(sizeof(float) * BATCH * 16  * LPTS);
  float* vbuf   = (float*)alloc(sizeof(float) * BATCH * 128 * LPTS);
  float* attn   = (float*)alloc(sizeof(float) * BATCH * 128 * LPTS);
  _Float16* qh  = (_Float16*)alloc(sizeof(_Float16) * (size_t)BATCH * LPTS * 32);
  _Float16* kh  = (_Float16*)alloc(sizeof(_Float16) * (size_t)BATCH * LPTS * 32);
  _Float16* vh  = (_Float16*)alloc(sizeof(_Float16) * (size_t)BATCH * 128 * LPTS);
  float* stats  = (float*)alloc(sizeof(float) * 2 * BATCH);
  float* pooled = (float*)alloc(sizeof(float) * BATCH * 256);
  float* fA     = (float*)alloc(sizeof(float) * BATCH * 128);
  float* fB     = (float*)alloc(sizeof(float) * BATCH * 64);
  float* fC     = (float*)alloc(sizeof(float) * BATCH * 64);
  float* trans  = (float*)alloc(sizeof(float) * BATCH * 36);

  auto conv = [&](const float* in, int Wi, int Bi, float* out_, int C, int O) {
    dim3 g(cdiv((size_t)O * LPTS, 256), BATCH);
    conv1x1_kernel<<<g, 256, 0, stream>>>(in, PARAM(Wi), PARAM(Bi), out_, C, O);
  };
  auto gn = [&](float* buf, int C, int L, int Gi, int Bi) {
    gn_stats_kernel<<<BATCH, 256, 0, stream>>>(buf, stats, C * L);
    size_t tot = (size_t)BATCH * C * L;
    gn_apply_kernel<<<cdiv(tot, 256), 256, 0, stream>>>(buf, stats, PARAM(Gi), PARAM(Bi),
                                                        C, L, 1);
  };
  auto fc = [&](const float* in, int Wi, int Bi, float* out_, int F, int O) {
    fc_kernel<<<cdiv((size_t)BATCH * O, 128), 128, 0, stream>>>(in, PARAM(Wi), PARAM(Bi),
                                                                out_, F, O);
  };

  // ---- normalize ----
  normalize_kernel<<<BATCH, 256, 0, stream>>>(X, xT);

  // ---- TNet ----
  conv(xT, T_C1W, T_C1B, h1, 6, 32);    gn(h1, 32, LPTS, T_N1G, T_N1B);
  conv(h1, T_C2W, T_C2B, h2, 32, 64);   gn(h2, 64, LPTS, T_N2G, T_N2B);
  conv(h2, T_C3W, T_C3B, h3, 64, 256);  gn(h3, 256, LPTS, T_N3G, T_N3B);
  maxpool_kernel<<<BATCH * 256, 256, 0, stream>>>(h3, pooled);
  fc(pooled, T_F1W, T_F1B, fA, 256, 128);  gn(fA, 128, 1, T_N4G, T_N4B);
  fc(fA, T_F2W, T_F2B, fB, 128, 64);       gn(fB, 64, 1, T_N5G, T_N5B);
  fc(fB, T_F3W, T_F3B, fC, 64, 36);
  tnet_final_kernel<<<1, 256, 0, stream>>>(fC, trans);
  apply_trans_kernel<<<cdiv((size_t)BATCH * DIN * LPTS, 256), 256, 0, stream>>>(trans, xT, x2);

  // ---- trunk ----
  conv(x2, C1W, C1B, h1, 6, 32);    gn(h1, 32, LPTS, N1G, N1B);
  conv(h1, C2W, C2B, h2, 32, 64);   gn(h2, 64, LPTS, N2G, N2B);
  conv(h2, C3W, C3B, h3, 64, 128);  gn(h3, 128, LPTS, N3G, N3B);   // h3 = attention input x

  // ---- attention (WMMA flash) ----
  conv(h3, A_QW, A_QB, qbuf, 128, 16);
  conv(h3, A_KW, A_KB, kbuf, 128, 16);
  conv(h3, A_VW, A_VB, vbuf, 128, 128);
  pack_qk_kernel<<<cdiv((size_t)BATCH * LPTS * 32, 256), 256, 0, stream>>>(qbuf, qh);
  pack_qk_kernel<<<cdiv((size_t)BATCH * LPTS * 32, 256), 256, 0, stream>>>(kbuf, kh);
  pack_v_kernel<<<cdiv((size_t)BATCH * 128 * LPTS, 256), 256, 0, stream>>>(vbuf, vh);
  {
    dim3 g(LPTS / 16 / 4, BATCH);   // 512 query tiles, 4 waves/block
    attn_flash_kernel<<<g, 128, 0, stream>>>(qh, kh, vh, h3, PARAM(A_GAMMA), attn);
  }

  // ---- head ----
  maxpool_kernel<<<BATCH * 128, 256, 0, stream>>>(attn, pooled);
  fc(pooled, F1W, F1B, fB, 128, 64);  gn(fB, 64, 1, N4G, N4B);
  fc(fB, F2W, F2B, fC, 64, 32);       gn(fC, 32, 1, N5G, N5B);
  fc_clip_kernel<<<1, 64, 0, stream>>>(fC, PARAM(F3W), PARAM(F3B), (float*)d_out, 32);
  #undef PARAM
}
